// TimeDomainDecodingLoss_63926293233693
// MI455X (gfx1250) — matrix operand order repaired and checked by reading, not compiled, not run on hardware
//
#include <hip/hip_runtime.h>

typedef __attribute__((ext_vector_type(16))) _Float16 v16h;
typedef __attribute__((ext_vector_type(8)))  float    v8f;

#define WIN   1024
#define NCLS  16
#define PI_F  3.14159265358979323846f

// XOR swizzle for the f16 log-mag matrix: row r, col k stored at r*1024 + (k ^ ((r&15)<<2)).
// Bank = ((k ^ swz) >> 1) & 63 -> 16 rows at same k hit 16 distinct banks (A-load phase).
#define LSWZ(r) (((r) & 15) << 2)

__constant__ int c_delays[NCLS] = {64, 92, 120, 148, 176, 204, 232, 260,
                                   288, 316, 344, 372, 400, 428, 456, 484};

__global__ void zero_ws_kernel(float* ws, int n) {
    int i = blockIdx.x * blockDim.x + threadIdx.x;
    if (i < n) ws[i] = 0.0f;
}

// One workgroup = 16 windows. LDS = exactly 160KB -> 2 workgroups / 320KB WGP.
__global__ __launch_bounds__(256)
void cepstral_loss_kernel(const float* __restrict__ audio,
                          const int*   __restrict__ gt,      // flat [total_windows]
                          float* __restrict__ ws,            // [0..nsamp-1] correct, [nsamp] CE sum
                          int wins_per_sample, int nsamp)
{
    __shared__ float    sRe[16 * WIN];      // 64KB
    __shared__ float    sIm[16 * WIN];      // 64KB
    __shared__ _Float16 sL [16 * WIN];      // 32KB  (log-mag A matrix, XOR-swizzled)
    float* sC = sRe;                        // reuse sRe for 8 partial C tiles (8*256 floats)

    const int tid = threadIdx.x;
    const long base = (long)blockIdx.x * 16 * WIN;

    // ---- load 16 windows as float4 (global_load_b128 / ds_store_b128), imag = 0 ----
    const float4 z4 = make_float4(0.f, 0.f, 0.f, 0.f);
    for (int idx = tid; idx < 16 * WIN / 4; idx += 256) {
        float4 v = ((const float4*)audio)[base / 4 + idx];
        ((float4*)sRe)[idx] = v;
        ((float4*)sIm)[idx] = z4;
    }
    __syncthreads();

    // ---- radix-2 DIF FFT, 10 stages, bit-reversed output order ----
    for (int s = 0; s < 10; ++s) {
        const int half = 512 >> s;
        for (int idx = tid; idx < 16 * 512; idx += 256) {
            int w = idx >> 9, t = idx & 511;
            int j = t & (half - 1);
            int g = t >> (9 - s);
            int i0 = (w << 10) + (g << (10 - s)) + j;
            int i1 = i0 + half;
            float ar = sRe[i0], ai = sIm[i0];
            float br = sRe[i1], bi = sIm[i1];
            float dr = ar - br, di = ai - bi;
            sRe[i0] = ar + br;
            sIm[i0] = ai + bi;
            float ang = (float)(j << s) * (-PI_F / 512.0f);  // W_N^{j*2^s}
            float sn, cs;
            __sincosf(ang, &sn, &cs);
            sRe[i1] = dr * cs - di * sn;
            sIm[i1] = dr * sn + di * cs;
        }
        __syncthreads();
    }

    // ---- log magnitude -> f16 A matrix (slot n holds bin bitrev10(n); B folds that) ----
    for (int idx = tid; idx < 16 * WIN; idx += 256) {
        int w = idx >> 10, n = idx & 1023;
        float re = sRe[idx], im = sIm[idx];
        float mag = __fsqrt_rn(re * re + im * im);
        sL[(w << 10) + (n ^ LSWZ(w))] = (_Float16)__logf(mag + 1e-5f);
    }
    __syncthreads();

    // ---- WMMA: cep[16 windows][16 delays] = A(16x1024) * B(1024x16), K split over 8 waves ----
    const int wave    = tid >> 5;
    const int lane    = tid & 31;
    const int nIdx    = lane & 15;   // N (delay) / M (window row) position within half-wave
    const int halfSel = lane >> 4;

    v8f c = {};
    const int dly = c_delays[nIdx];
    #pragma unroll
    for (int kk4 = 0; kk4 < 4; ++kk4) {          // unrolled: 4 static v_wmma, cos work overlaps
        const int kbase = (wave * 4 + kk4) * 32;
        v16h a, b;
        // A layout (ISA 16-bit A 16x32): lanes0-15 M=lane; VGPR r: K=2r,2r+1 (+8 hi half); +16 for VGPR4-7
        #pragma unroll
        for (int e = 0; e < 16; ++e) {
            int k = ((e >> 3) << 4) + (halfSel << 3) + (e & 7);
            a[e] = sL[(nIdx << 10) + ((kbase + k) ^ LSWZ(nIdx))];
        }
        // B layout (matches A): K = 8*half + (e&7) for e<8, 16 + 8*half + (e&7) for e>=8, N=lane&15.
        // Weight = cos(2*pi*bitrev10(slot)*delay/1024)/1024 -> folds irfft + bit-reversal.
        #pragma unroll
        for (int e = 0; e < 16; ++e) {
            int k  = kbase + ((e >> 3) << 4) + (halfSel << 3) + (e & 7);
            int kr = (int)(__brev((unsigned)k) >> 22);
            float ang = (float)((kr * dly) & 1023) * (PI_F / 512.0f);
            b[e] = (_Float16)(__cosf(ang) * (1.0f / 1024.0f));
        }
        c = __builtin_amdgcn_wmma_f32_16x16x32_f16(false, a, false, b,
                                                   (short)0, c, false, false);
    }
    // stash partial C tiles; sum over the 8 waves that split K
    {
        float* myC = sC + wave * 256;
        #pragma unroll
        for (int r = 0; r < 8; ++r) myC[r * 32 + lane] = c[r];
    }
    __syncthreads();

    // ---- epilogue: wave 0 handles all 16 windows of this tile ----
    if (wave == 0) {
        #pragma unroll
        for (int r = 0; r < 8; ++r) {
            float v = 0.0f;
            #pragma unroll
            for (int wv = 0; wv < 8; ++wv) v += sC[wv * 256 + r * 32 + lane];
            // C layout: N = lane&15, window m = r + 8*halfSel
            float mx = v;
            #pragma unroll
            for (int m = 8; m; m >>= 1) mx = fmaxf(mx, __shfl_xor(mx, m, 32));
            float se = __expf(v - mx);
            #pragma unroll
            for (int m = 8; m; m >>= 1) se += __shfl_xor(se, m, 32);
            float lse = mx + __logf(se);
            int am = (v == mx) ? nIdx : 99;          // BETA=1e10 softmax*range == argmax
            #pragma unroll
            for (int m = 8; m; m >>= 1) { int o = __shfl_xor(am, m, 32); am = (o < am) ? o : am; }

            int gw = blockIdx.x * 16 + r + 8 * halfSel;
            int g  = gt[gw];
            float vg = __shfl(v, (halfSel << 4) + g, 32);  // cep value at gt class

            if (nIdx == 0) {
                // +0.0001 logit shift cancels in log_softmax
                float ce  = lse - vg;
                float cor = (am == g) ? 1.0f : 0.0f;
                atomicAdd(&ws[nsamp], ce);
                atomicAdd(&ws[gw / wins_per_sample], cor);
            }
        }
    }
}

__global__ void finalize_kernel(const float* __restrict__ ws,
                                const int* __restrict__ nr,
                                const int* __restrict__ rv,
                                float* __restrict__ out,
                                int nsamp, int wins_per_sample)
{
    int l = threadIdx.x;  // 32 threads (one wave)
    float cnt = (l < nsamp) ? ws[l]        : 0.0f;
    float nrf = (l < nsamp) ? (float)nr[l] : 0.0f;
    float rvf = (l < nsamp) ? (float)rv[l] : 0.0f;
    float nerr  = fmaxf((float)wins_per_sample - cnt, 0.0f);
    float denom = fmaxf(rvf, 1.0f);
    float lb = (l < nsamp) ? (nerr / denom) : 0.0f;

    float scnt = cnt, snr = nrf, srv = rvf, slb = lb;
    #pragma unroll
    for (int m = 16; m; m >>= 1) {
        scnt += __shfl_xor(scnt, m, 32);
        snr  += __shfl_xor(snr,  m, 32);
        srv  += __shfl_xor(srv,  m, 32);
        slb  += __shfl_xor(slb,  m, 32);
    }
    if (l == 0) {
        float total = (float)nsamp * (float)wins_per_sample;
        out[0] = ws[nsamp] / total;          // sym_err_loss
        out[1] = 1.0f - scnt / total;        // sym_err_rate
        out[2] = slb / (float)nsamp;         // avg_err_reduction_loss
        out[3] = snr / total;                // no_reverb_sym_err_rate
        out[4] = srv / total;                // reverb_sym_err_rate
    }
}

extern "C" void kernel_launch(void* const* d_in, const int* in_sizes, int n_in,
                              void* d_out, int out_size, void* d_ws, size_t ws_size,
                              hipStream_t stream) {
    const float* audio = (const float*)d_in[0];
    const int*   gt    = (const int*)d_in[1];
    const int*   nr    = (const int*)d_in[2];
    const int*   rv    = (const int*)d_in[3];
    float* out = (float*)d_out;
    float* ws  = (float*)d_ws;

    const int nsamp           = in_sizes[2];          // 32
    const int total_wins      = in_sizes[1];          // 16384
    const int wins_per_sample = total_wins / nsamp;   // 512
    const int nblocks         = total_wins / 16;      // 1024 tiles of 16 windows

    zero_ws_kernel<<<1, 64, 0, stream>>>(ws, nsamp + 1);
    cepstral_loss_kernel<<<nblocks, 256, 0, stream>>>(audio, gt, ws, wins_per_sample, nsamp);
    finalize_kernel<<<1, 32, 0, stream>>>(ws, nr, rv, out, nsamp, wins_per_sample);
}
